// MoLoRA_5488968204634
// MI455X (gfx1250) — matrix operand order repaired and checked by reading, not compiled, not run on hardware
//
#include <hip/hip_runtime.h>

typedef __attribute__((ext_vector_type(8)))  float        v8f;
typedef __attribute__((ext_vector_type(16))) __bf16       v16bf;
typedef __attribute__((ext_vector_type(8)))  __bf16       v8bf;
typedef __attribute__((ext_vector_type(4)))  unsigned int v4u;
typedef __attribute__((ext_vector_type(4)))  int          v4i;
typedef __attribute__((ext_vector_type(8)))  int          v8i;

#define TOKENS 16384
#define DDIM   2048
#define HDIM   256
#define NEXP   5
#define RANK   16
#define ER     80     // E*R
#define ERP    96     // padded to 6 tiles of 16
#define TILE_M 64
#define KC     32     // K per chunk == WMMA K
#define NB3    128    // N-chunk of phase 3b
#define SCALING 2.0f

// ---- bf16 split helpers (manual RNE; x ~= hi + lo, lo = bf16(x - hi)) ----
__device__ __forceinline__ unsigned short f2bf(float f) {
  unsigned int u = __float_as_uint(f);
  unsigned int r = u + 0x7fffu + ((u >> 16) & 1u);
  return (unsigned short)(r >> 16);
}
__device__ __forceinline__ float bf2f(unsigned short u) {
  return __uint_as_float(((unsigned int)u) << 16);
}

__device__ __forceinline__ v8f wmma_bf16(v16bf a, v16bf b, v8f c) {
  return __builtin_amdgcn_wmma_f32_16x16x32_bf16(false, a, false, b, (short)0, c, false, false);
}

// A-fragment (16x32 bf16): lane half selects K {h*16..h*16+15}
__device__ __forceinline__ v16bf frag_a(const unsigned short* rowbase, int half) {
  return *(const v16bf*)(rowbase + half * 16);
}
// B-fragment (32x16 bf16): lane needs 16 contiguous K at its column
__device__ __forceinline__ v16bf frag_b(const unsigned short* p) {
  return *(const v16bf*)p;
}

// ---- Tensor Data Mover: 2D tile (rows x cols bf16) -> LDS, count=1, type=2 ----
__device__ __forceinline__ void tdm_load_2d_bf16(unsigned lds_off, const void* gptr,
                                                 unsigned rows, unsigned cols,
                                                 unsigned stride_elems) {
  unsigned long long ga = (unsigned long long)(uintptr_t)gptr;
  v4u g0;
  g0[0] = 1u;                                           // count=1, user mode
  g0[1] = lds_off;                                      // lds_addr (bytes)
  g0[2] = (unsigned)(ga & 0xffffffffu);                 // global_addr[31:0]
  g0[3] = (unsigned)((ga >> 32) & 0x01ffffffu) | 0x80000000u; // addr[56:32] | type=2
  v8i g1;
  g1[0] = (int)(1u << 16);                              // wgmask=0, data_size=1 (2B)
  g1[1] = (int)((cols & 0xffffu) << 16);                // tensor_dim0[15:0] @ bits 63:48
  g1[2] = (int)(((cols >> 16) & 0xffffu) | ((rows & 0xffffu) << 16)); // dim0 hi | dim1 lo
  g1[3] = (int)(((rows >> 16) & 0xffffu) | ((cols & 0xffffu) << 16)); // dim1 hi | tile_dim0
  g1[4] = (int)(rows & 0xffffu);                        // tile_dim1 | tile_dim2=0
  g1[5] = (int)stride_elems;                            // tensor_dim0_stride[31:0]
  g1[6] = 0;
  g1[7] = 0;
  v4i z4 = {0, 0, 0, 0};
#if __clang_major__ >= 23
  v8i z8 = {0, 0, 0, 0, 0, 0, 0, 0};
  __builtin_amdgcn_tensor_load_to_lds(g0, g1, z4, z4, z8, 0);
#else
  __builtin_amdgcn_tensor_load_to_lds(g0, g1, z4, z4, 0);
#endif
}

// ======== one-time weight conversion: bf16 hi/lo, transposed for LDS staging ====
// W1hT/W1lT: [H=256][D=2048]   ATh/ATl: [ERP=96][D]   BTh/BTl: [D][ERP]
__global__ __launch_bounds__(256)
void molora_precvt(const float* __restrict__ Am, const float* __restrict__ Bm,
                   const float* __restrict__ W1,
                   unsigned short* __restrict__ W1hT, unsigned short* __restrict__ W1lT,
                   unsigned short* __restrict__ ATh,  unsigned short* __restrict__ ATl,
                   unsigned short* __restrict__ BTh,  unsigned short* __restrict__ BTl)
{
  int idx = blockIdx.x * 256 + threadIdx.x;
  if (idx < DDIM * HDIM) {
    int d = idx / HDIM, h = idx % HDIM;
    float v = W1[idx];
    unsigned short hb = f2bf(v);
    W1hT[(size_t)h * DDIM + d] = hb;
    W1lT[(size_t)h * DDIM + d] = f2bf(v - bf2f(hb));
  } else if (idx < DDIM * HDIM + ERP * DDIM) {
    int j = idx - DDIM * HDIM;
    int c = j / DDIM, d = j % DDIM;
    float v = 0.0f;
    if (c < ER) v = Am[(size_t)(c >> 4) * DDIM * RANK + (size_t)d * RANK + (c & 15)];
    unsigned short hb = f2bf(v);
    ATh[(size_t)c * DDIM + d] = hb;
    ATl[(size_t)c * DDIM + d] = f2bf(v - bf2f(hb));
  } else {
    int j = idx - DDIM * HDIM - ERP * DDIM;
    int n = j / ERP, c = j % ERP;
    float v = 0.0f;
    if (c < ER) v = Bm[(size_t)(c >> 4) * RANK * DDIM + (size_t)(c & 15) * DDIM + n];
    unsigned short hb = f2bf(v);
    BTh[(size_t)n * ERP + c] = hb;
    BTl[(size_t)n * ERP + c] = f2bf(v - bf2f(hb));
  }
}

// ---- LDS pool, 63488 B ----
// Region A (K loop):  xh@0(4K) xl@4096(4K) w1h@8192(16K) w1l@24576(16K)
//                     ath@40960(6K) atl@47104(6K)                 -> 53248
// Persistent:         b1l@53248(1K) w2l@54272(5K) part@59392(2.5K) wsb@61952(1.5K)
// Region B (phase 3): lowSh@0(12K) lowSl@12288(12K)  then (after frag load)
//                     bchTh@0(24K) bchTl@24576(24K)  [TDM destinations]

__global__ __launch_bounds__(256)
void molora_fused(const float* __restrict__ x,
                  const float* __restrict__ base,
                  const float* __restrict__ b1,
                  const float* __restrict__ W2,
                  const float* __restrict__ b2,
                  const unsigned short* __restrict__ W1hT,
                  const unsigned short* __restrict__ W1lT,
                  const unsigned short* __restrict__ ATh,
                  const unsigned short* __restrict__ ATl,
                  const unsigned short* __restrict__ BTh,
                  const unsigned short* __restrict__ BTl,
                  float* __restrict__ out)
{
  __shared__ alignas(64) char pool[63488];
  unsigned short* xh  = (unsigned short*)(pool);
  unsigned short* xl  = (unsigned short*)(pool + 4096);
  unsigned short* w1h = (unsigned short*)(pool + 8192);
  unsigned short* w1l = (unsigned short*)(pool + 24576);
  unsigned short* ath = (unsigned short*)(pool + 40960);
  unsigned short* atl = (unsigned short*)(pool + 47104);
  float* b1l  = (float*)(pool + 53248);
  float* w2l  = (float*)(pool + 54272);
  float* part = (float*)(pool + 59392);
  float* wsb  = (float*)(pool + 61952);
  unsigned short* lowSh = (unsigned short*)(pool);
  unsigned short* lowSl = (unsigned short*)(pool + 12288);
  unsigned short* bchTh = (unsigned short*)(pool);
  unsigned short* bchTl = (unsigned short*)(pool + 24576);
  const unsigned ldsbase = (unsigned)(uintptr_t)pool;

  const int tid  = threadIdx.x;
  const int lane = tid & 31;
  const int wave = tid >> 5;
  const int wm   = wave & 3;      // rows wm*16..+15
  const int wn   = wave >> 2;     // col group 0/1
  const int half = lane >> 4;
  const int lrow = lane & 15;
  const int t0   = blockIdx.x * TILE_M;

  b1l[tid] = b1[tid];
  #pragma unroll
  for (int e = 0; e < NEXP; ++e) w2l[tid * NEXP + e] = W2[tid * NEXP + e];

  v8f accR[8];   // router: 16 x 128 per wave
  v8f accL[3];   // low:    16 x 48  per wave
  #pragma unroll
  for (int t = 0; t < 8; ++t) accR[t] = v8f{0,0,0,0,0,0,0,0};
  #pragma unroll
  for (int t = 0; t < 3; ++t) accL[t] = v8f{0,0,0,0,0,0,0,0};

  // ---------- Phase 1 + 3a: router GEMM + low = x@A_all (bf16x3, fp32 accum) ----
  // Weight slabs move by TDM (waves 0-3, one DMA each) while all threads do the
  // x hi/lo split in VALU; issuing waves s_wait_tensorcnt before the barrier.
  for (int k0 = 0; k0 < DDIM; k0 += KC) {
    if (k0 + KC < DDIM)
      __builtin_prefetch(x + (size_t)(t0 + (tid >> 2)) * DDIM + k0 + KC, 0, 0);
    __syncthreads();   // LDS free: previous chunk fully consumed

    if (wave == 0)      tdm_load_2d_bf16(ldsbase +  8192, W1hT + k0, HDIM, KC, DDIM);
    else if (wave == 1) tdm_load_2d_bf16(ldsbase + 24576, W1lT + k0, HDIM, KC, DDIM);
    else if (wave == 2) tdm_load_2d_bf16(ldsbase + 40960, ATh  + k0, ERP,  KC, DDIM);
    else if (wave == 3) tdm_load_2d_bf16(ldsbase + 47104, ATl  + k0, ERP,  KC, DDIM);

    { // x -> split bf16 hi/lo, [64][32]  (overlaps the weight DMAs)
      int r = tid >> 2, q = tid & 3;
      const float4* xs = (const float4*)(x + (size_t)(t0 + r) * DDIM + k0 + q * 8);
      float4 v0 = xs[0], v1 = xs[1];
      float f[8] = {v0.x, v0.y, v0.z, v0.w, v1.x, v1.y, v1.z, v1.w};
      unsigned int hw[4], lw[4];
      #pragma unroll
      for (int j = 0; j < 4; ++j) {
        unsigned short ha = f2bf(f[2*j]), hb = f2bf(f[2*j+1]);
        unsigned short la = f2bf(f[2*j]   - bf2f(ha));
        unsigned short lb = f2bf(f[2*j+1] - bf2f(hb));
        hw[j] = (unsigned)ha | ((unsigned)hb << 16);
        lw[j] = (unsigned)la | ((unsigned)lb << 16);
      }
      uint4* dh = (uint4*)(xh + r * KC + q * 8);
      uint4* dl = (uint4*)(xl + r * KC + q * 8);
      *dh = make_uint4(hw[0], hw[1], hw[2], hw[3]);
      *dl = make_uint4(lw[0], lw[1], lw[2], lw[3]);
    }
    if (wave < 4) __builtin_amdgcn_s_wait_tensorcnt(0);
    __syncthreads();   // all staging (DMA + x) visible

    v16bf ah = frag_a(xh + (wm * 16 + lrow) * KC, half);
    v16bf al = frag_a(xl + (wm * 16 + lrow) * KC, half);
    #pragma unroll
    for (int t = 0; t < 8; ++t) {
      int c = wn * 128 + t * 16 + lrow;
      v16bf bh = frag_b(w1h + c * KC + half * 16);
      v16bf bl = frag_b(w1l + c * KC + half * 16);
      accR[t] = wmma_bf16(ah, bh, accR[t]);   // hi*hi
      accR[t] = wmma_bf16(ah, bl, accR[t]);   // hi*lo
      accR[t] = wmma_bf16(al, bh, accR[t]);   // lo*hi
    }
    #pragma unroll
    for (int t = 0; t < 3; ++t) {
      int c = wn * 48 + t * 16 + lrow;
      v16bf bh = frag_b(ath + c * KC + half * 16);
      v16bf bl = frag_b(atl + c * KC + half * 16);
      accL[t] = wmma_bf16(ah, bh, accL[t]);
      accL[t] = wmma_bf16(ah, bl, accL[t]);
      accL[t] = wmma_bf16(al, bh, accL[t]);
    }
  }

  // ---------- Phase 2: SiLU -> W2 -> softmax -> top2 -> weights ------------------
  float p[8][NEXP];
  #pragma unroll
  for (int r = 0; r < 8; ++r)
    #pragma unroll
    for (int e = 0; e < NEXP; ++e) p[r][e] = 0.0f;

  #pragma unroll
  for (int t = 0; t < 8; ++t) {
    int c = wn * 128 + t * 16 + lrow;
    float bb = b1l[c];
    float w2c[NEXP];
    #pragma unroll
    for (int e = 0; e < NEXP; ++e) w2c[e] = w2l[c * NEXP + e];
    #pragma unroll
    for (int r = 0; r < 8; ++r) {
      float z = accR[t][r] + bb;
      float s = z / (1.0f + __expf(-z));
      #pragma unroll
      for (int e = 0; e < NEXP; ++e) p[r][e] += s * w2c[e];
    }
  }
  #pragma unroll
  for (int r = 0; r < 8; ++r) {
    #pragma unroll
    for (int e = 0; e < NEXP; ++e) {
      float v = p[r][e];
      v += __shfl_xor(v, 1, 32);
      v += __shfl_xor(v, 2, 32);
      v += __shfl_xor(v, 4, 32);
      v += __shfl_xor(v, 8, 32);
      if (lrow == 0) part[(wn * 64 + wm * 16 + r + half * 8) * NEXP + e] = v;
    }
  }
  __syncthreads();
  if (tid < TILE_M) {
    float l[NEXP], w[NEXP];
    float mx = -1e30f;
    #pragma unroll
    for (int e = 0; e < NEXP; ++e) {
      l[e] = part[tid * NEXP + e] + part[(64 + tid) * NEXP + e] + b2[e];
      mx = fmaxf(mx, l[e]);
    }
    #pragma unroll
    for (int e = 0; e < NEXP; ++e) w[e] = __expf(l[e] - mx);
    int i1 = 0;
    #pragma unroll
    for (int e = 1; e < NEXP; ++e) if (w[e] > w[i1]) i1 = e;
    int i2 = (i1 == 0) ? 1 : 0;
    #pragma unroll
    for (int e = 0; e < NEXP; ++e) if (e != i1 && w[e] > w[i2]) i2 = e;
    float den = w[i1] + w[i2];
    #pragma unroll
    for (int e = 0; e < 6; ++e) wsb[tid * 6 + e] = 0.0f;
    wsb[tid * 6 + i1] = SCALING * w[i1] / den;
    wsb[tid * 6 + i2] = SCALING * w[i2] / den;
  }
  __syncthreads();

  // weighted low -> bf16 hi/lo in LDS (A-matrix of phase 3b)
  #pragma unroll
  for (int t = 0; t < 3; ++t) {
    int e = wn * 3 + t;              // one expert per 16-col tile (e==5 -> w=0 pad)
    int c = wn * 48 + t * 16 + lrow;
    #pragma unroll
    for (int r = 0; r < 8; ++r) {
      int row = wm * 16 + r + half * 8;
      float lf = accL[t][r] * wsb[row * 6 + e];
      unsigned short hb = f2bf(lf);
      lowSh[row * ERP + c] = hb;
      lowSl[row * ERP + c] = f2bf(lf - bf2f(hb));
    }
  }
  __syncthreads();

  // pull this wave's 3b A-fragments into registers, then free the LDS for TDM
  v16bf aoh[3], aol[3];
  {
    const unsigned short* lsh = lowSh + (wm * 16 + lrow) * ERP;
    const unsigned short* lsl = lowSl + (wm * 16 + lrow) * ERP;
    #pragma unroll
    for (int ki = 0; ki < 3; ++ki) {
      aoh[ki] = frag_a(lsh + ki * KC, half);
      aol[ki] = frag_a(lsl + ki * KC, half);
    }
  }
  __syncthreads();

  // ---------- Phase 3b: out = lowS[64x96] @ B_all[96x2048] + base ----------------
  for (int nc = 0; nc < DDIM / NB3; ++nc) {
    const int nb = nc * NB3;
    if (wave == 0)      tdm_load_2d_bf16(ldsbase,         BTh + (size_t)nb * ERP, NB3, ERP, ERP);
    else if (wave == 1) tdm_load_2d_bf16(ldsbase + 24576, BTl + (size_t)nb * ERP, NB3, ERP, ERP);
    if (wave < 2) __builtin_amdgcn_s_wait_tensorcnt(0);
    __syncthreads();

    v8f accO[4];
    #pragma unroll
    for (int t = 0; t < 4; ++t) accO[t] = v8f{0,0,0,0,0,0,0,0};
    #pragma unroll
    for (int ki = 0; ki < 3; ++ki) {
      const int kk = ki * KC;
      #pragma unroll
      for (int t = 0; t < 4; ++t) {
        int c = wn * 64 + t * 16 + lrow;
        v16bf bh = frag_b(bchTh + c * ERP + kk + half * 16);
        v16bf bl = frag_b(bchTl + c * ERP + kk + half * 16);
        accO[t] = wmma_bf16(aoh[ki], bh, accO[t]);
        accO[t] = wmma_bf16(aol[ki], bh, accO[t]);
        accO[t] = wmma_bf16(aoh[ki], bl, accO[t]);
      }
    }
    #pragma unroll
    for (int t = 0; t < 4; ++t) {
      int c = nb + wn * 64 + t * 16 + lrow;
      #pragma unroll
      for (int r = 0; r < 8; ++r) {
        int row = t0 + wm * 16 + r + half * 8;
        size_t idx = (size_t)row * DDIM + c;
        out[idx] = base[idx] + accO[t][r];
      }
    }
    __syncthreads();   // bchT reuse fence
  }
}

extern "C" void kernel_launch(void* const* d_in, const int* in_sizes, int n_in,
                              void* d_out, int out_size, void* d_ws, size_t ws_size,
                              hipStream_t stream) {
  const float* x    = (const float*)d_in[0];
  const float* base = (const float*)d_in[1];
  const float* A    = (const float*)d_in[2];
  const float* B    = (const float*)d_in[3];
  const float* W1   = (const float*)d_in[4];
  const float* b1   = (const float*)d_in[5];
  const float* W2   = (const float*)d_in[6];
  const float* b2   = (const float*)d_in[7];
  float* out = (float*)d_out;
  (void)in_sizes; (void)n_in; (void)out_size; (void)ws_size;

  char* ws = (char*)d_ws;   // needs 3670016 B
  unsigned short* W1hT = (unsigned short*)(ws + 0);
  unsigned short* W1lT = (unsigned short*)(ws + 1048576);
  unsigned short* ATh  = (unsigned short*)(ws + 2097152);
  unsigned short* ATl  = (unsigned short*)(ws + 2490368);
  unsigned short* BTh  = (unsigned short*)(ws + 2883584);
  unsigned short* BTl  = (unsigned short*)(ws + 3276800);

  // one-time bf16 hi/lo split + transpose of weights (L2-resident afterwards)
  hipLaunchKernelGGL(molora_precvt, dim3(3584), dim3(256), 0, stream,
                     A, B, W1, W1hT, W1lT, ATh, ATl, BTh, BTl);
  // fused router + MoLoRA
  hipLaunchKernelGGL(molora_fused, dim3(TOKENS / TILE_M), dim3(256), 0, stream,
                     x, base, b1, W2, b2, W1hT, W1lT, ATh, ATl, BTh, BTl, out);
}
